// CrossAtt_10960756540399
// MI455X (gfx1250) — matrix operand order repaired and checked by reading, not compiled, not run on hardware
//
#include <hip/hip_runtime.h>
#include <cstdint>

typedef __attribute__((ext_vector_type(16))) _Float16 v16h;
typedef __attribute__((ext_vector_type(8)))  float    v8f;
typedef __attribute__((ext_vector_type(4)))  uint32_t u32x4;

#define NPIX 4096
#define CIN  256
#define DV   128
#define DQK  16
#define QKPAD 32   // q/k rows padded to 32 halves (d=16..31 are stored zeros)

union AFrag { v16h v; uint32_t u[8]; };

__device__ __forceinline__ v8f wmma16x16x32(v16h a, v16h b, v8f c) {
  // 8 args: (neg_a, A, neg_b, B, c_mod, C, reuse_a, reuse_b)
  return __builtin_amdgcn_wmma_f32_16x16x32_f16(false, a, false, b, (short)0, c,
                                                false, false);
}

__device__ __forceinline__ v8f zero8() {
  v8f z;
#pragma unroll
  for (int i = 0; i < 8; ++i) z[i] = 0.0f;
  return z;
}

// A-fragment (16-bit A 16x32): lane-half `hi`, VGPR j covers halves
// [hi*16 + 2j, +1].  j0..3 -> 16B at base+hi*16,
// j4..7 -> 16B at base+hi*16+8.  `base` = row start (32B aligned).
__device__ __forceinline__ void load_afrag(AFrag& f, const _Float16* base, int hi) {
  u32x4 a = *(const u32x4*)(base + hi * 16);
  u32x4 b = *(const u32x4*)(base + hi * 16 + 8);
#pragma unroll
  for (int j = 0; j < 4; ++j) { f.u[j] = a[j]; f.u[j + 4] = b[j]; }
}

// B-fragment (32x16, 16-bit): lane-half `hi`, VGPR j covers halves
// col[hi*16 + 2j .. +1] of a 32-half contiguous K-column.  Two 16B loads.
__device__ __forceinline__ void load_bfrag(AFrag& f, const _Float16* col, int hi) {
  u32x4 a = *(const u32x4*)(col + hi * 16);
  u32x4 b = *(const u32x4*)(col + hi * 16 + 8);
#pragma unroll
  for (int j = 0; j < 4; ++j) { f.u[j] = a[j]; f.u[j + 4] = b[j]; }
}

// ---------------------------------------------------------------------------
// Kernel 1: Wcat (O,C,3,3) f32 -> Wh[tap][o][c] f16
// ---------------------------------------------------------------------------
__global__ __launch_bounds__(256) void prep_w_kernel(const float* __restrict__ Wcat,
                                                     _Float16* __restrict__ Wh) {
  int idx = blockIdx.x * 256 + threadIdx.x;   // (tap,o,c)
  if (idx >= 9 * 256 * 256) return;
  int c = idx & 255, o = (idx >> 8) & 255, tap = idx >> 16;
  int kh = tap / 3, kw = tap % 3;
  Wh[idx] = (_Float16)Wcat[((o * 256 + c) * 3 + kh) * 3 + kw];
}

// ---------------------------------------------------------------------------
// Kernel 2: QKV projections (LDS-staged 1x1 conv).
//  Qh[slot][n][32] (pre-scaled by 0.25, halves 16..31 zeroed),
//  Kh[slot][n][32] (halves 16..31 zeroed), Vh[slot][c][n].  slot = input*2+b
// ---------------------------------------------------------------------------
__global__ __launch_bounds__(256) void qkv_kernel(
    const float* __restrict__ x1, const float* __restrict__ x2,
    const float* __restrict__ Wq, const float* __restrict__ bq,
    const float* __restrict__ Wk, const float* __restrict__ bk,
    const float* __restrict__ Wv, const float* __restrict__ bv,
    _Float16* __restrict__ Qh, _Float16* __restrict__ Kh,
    _Float16* __restrict__ Vh) {
  __shared__ float xs[CIN * 32];                  // [c][32 pixels]
  const int inp = blockIdx.z, b = blockIdx.y;
  const int ib = inp * 2 + b;
  const float* x = (inp == 0 ? x1 : x2) + (size_t)b * CIN * NPIX;
  const int n0 = blockIdx.x * 32;
  const int t = threadIdx.x;
#pragma unroll 4
  for (int i = 0; i < 32; ++i) {                  // 256c x 32n cooperative load
    int flat = t + 256 * i;
    int c = flat >> 5, n = flat & 31;
    xs[flat] = x[(size_t)c * NPIX + n0 + n];
  }
  __syncthreads();
#pragma unroll 1
  for (int i = 0; i < 20; ++i) {                  // 160 rows x 32 pixels
    int flat = t + 256 * i;
    int r = flat >> 5, n = flat & 31;             // r uniform per wave
    const float* Wrow;
    float bias;
    if (r < 16)      { Wrow = Wq + r * CIN;        bias = bq[r]; }
    else if (r < 32) { Wrow = Wk + (r - 16) * CIN; bias = bk[r - 16]; }
    else             { Wrow = Wv + (r - 32) * CIN; bias = bv[r - 32]; }
    float acc = 0.0f;
#pragma unroll 8
    for (int c = 0; c < CIN; ++c) acc += Wrow[c] * xs[c * 32 + n];
    acc += bias;
    int ng = n0 + n;
    if (r < 16) {
      size_t base = ((size_t)ib * NPIX + ng) * QKPAD;
      Qh[base + r]      = (_Float16)(acc * 0.25f);
      Qh[base + 16 + r] = (_Float16)0.0f;         // zero padding half
    } else if (r < 32) {
      size_t base = ((size_t)ib * NPIX + ng) * QKPAD;
      Kh[base + (r - 16)]      = (_Float16)acc;
      Kh[base + 16 + (r - 16)] = (_Float16)0.0f;  // zero padding half
    } else {
      Vh[((size_t)ib * DV + (r - 32)) * NPIX + ng] = (_Float16)acc;
    }
  }
}

// ---------------------------------------------------------------------------
// Kernel 3: streaming attention. One wave = one 16-query tile, full 128-row V.
//  2-pass softmax (pass1 max, pass2 exp/sum + WMMA accumulate).
//  Writes out_branch[c<128][n] = gamma*O + x_self.
// ---------------------------------------------------------------------------
__global__ __launch_bounds__(256) void attn_kernel(
    const float* __restrict__ x1, const float* __restrict__ x2,
    const float* __restrict__ gammap,
    const _Float16* __restrict__ Qh, const _Float16* __restrict__ Kh,
    const _Float16* __restrict__ Vh,
    float* __restrict__ out1, float* __restrict__ out2) {
  __shared__ _Float16 Pbuf[8][16 * 32];           // per-wave [q][k] tile
  __shared__ float    ssum[8][16];
  const int w = threadIdx.x >> 5, lane = threadIdx.x & 31;
  const int r = lane & 15, hi = lane >> 4;
  const int z = blockIdx.z, b = blockIdx.y;
  const int q0 = blockIdx.x * 128 + w * 16;
  const int self_slot = z * 2 + b, other_slot = (1 - z) * 2 + b;
  const _Float16* Qs = Qh + (size_t)self_slot * NPIX * QKPAD;
  const _Float16* Ko = Kh + (size_t)other_slot * NPIX * QKPAD;
  const _Float16* Vs = Vh + (size_t)self_slot * DV * NPIX;
  const float* xself = (z == 0 ? x1 : x2) + (size_t)b * CIN * NPIX;
  float* outb = (z == 0 ? out1 : out2) + (size_t)b * CIN * NPIX;
  _Float16* Pw = &Pbuf[w][0];

  // Q A-fragment: M=q, K=d; rows are zero-padded so the hi=1 lane half loads real zeros.
  AFrag qa;
  load_afrag(qa, Qs + (size_t)(q0 + r) * QKPAD, hi);

  // ---- pass 1: row maxima ----
  float rmax[8];
#pragma unroll
  for (int i = 0; i < 8; ++i) rmax[i] = -1e30f;
#pragma unroll 1
  for (int m0 = 0; m0 < NPIX; m0 += 32) {
    AFrag kb0, kb1;                                // B: K=d, N=key; hi=1 half
    load_bfrag(kb0, Ko + (size_t)(m0 + r) * QKPAD, hi);        // reads zeros
    load_bfrag(kb1, Ko + (size_t)(m0 + 16 + r) * QKPAD, hi);
    v8f s0 = wmma16x16x32(qa.v, kb0.v, zero8());
    v8f s1 = wmma16x16x32(qa.v, kb1.v, zero8());
#pragma unroll
    for (int i = 0; i < 8; ++i)
      rmax[i] = fmaxf(rmax[i], fmaxf(s0[i], s1[i]));
  }
#pragma unroll
  for (int i = 0; i < 8; ++i) {
#pragma unroll
    for (int d = 1; d < 16; d <<= 1)
      rmax[i] = fmaxf(rmax[i], __shfl_xor(rmax[i], d, 32));
  }

  // ---- pass 2: exp / sum / O accumulation ----
  float lsum[8];
#pragma unroll
  for (int i = 0; i < 8; ++i) lsum[i] = 0.0f;
  v8f O[8];
#pragma unroll
  for (int t = 0; t < 8; ++t) O[t] = zero8();

#pragma unroll 1
  for (int m0 = 0; m0 < NPIX; m0 += 32) {
    if (m0 + 32 < NPIX)                            // prefetch next V block
      __builtin_prefetch(Vs + (size_t)r * NPIX + m0 + 32, 0, 1);
    AFrag kb0, kb1;
    load_bfrag(kb0, Ko + (size_t)(m0 + r) * QKPAD, hi);
    load_bfrag(kb1, Ko + (size_t)(m0 + 16 + r) * QKPAD, hi);
    v8f s0 = wmma16x16x32(qa.v, kb0.v, zero8());
    v8f s1 = wmma16x16x32(qa.v, kb1.v, zero8());
#pragma unroll
    for (int i = 0; i < 8; ++i) {                  // element (q=i+hi*8, k)
      float p0 = __expf(s0[i] - rmax[i]);
      float p1 = __expf(s1[i] - rmax[i]);
      lsum[i] += p0 + p1;
      Pw[(i + hi * 8) * 32 + r]      = (_Float16)p0;   // P[q][k]
      Pw[(i + hi * 8) * 32 + 16 + r] = (_Float16)p1;
    }
    asm volatile("s_wait_dscnt 0x0" ::: "memory");     // wave-local LDS RAW
    AFrag pb;                                          // B: K=k, N=q
    load_bfrag(pb, Pw + r * 32, hi);                   // ds_load_b128 x2
#pragma unroll
    for (int t = 0; t < 8; ++t) {                      // A: V (16c x 32k)
      AFrag va;
      load_afrag(va, Vs + (size_t)(t * 16 + r) * NPIX + m0, hi);
      O[t] = wmma16x16x32(va.v, pb.v, O[t]);
    }
    asm volatile("" ::: "memory");                     // P tile stays live until read
  }

  // ---- epilogue: 1/sum (lane-layout transpose via LDS), gamma, residual ----
#pragma unroll
  for (int i = 0; i < 8; ++i) {
#pragma unroll
    for (int d = 1; d < 16; d <<= 1) lsum[i] += __shfl_xor(lsum[i], d, 32);
  }
  if (r == 0) {
#pragma unroll
    for (int i = 0; i < 8; ++i) ssum[w][i + hi * 8] = lsum[i];
  }
  asm volatile("s_wait_dscnt 0x0" ::: "memory");
  float invl = 1.0f / ssum[w][r];                      // q = r = N-col of O
  float g = gammap[0];
#pragma unroll
  for (int t = 0; t < 8; ++t) {
#pragma unroll
    for (int i = 0; i < 8; ++i) {
      int c = t * 16 + i + hi * 8;
      size_t idx = (size_t)c * NPIX + q0 + r;
      outb[idx] = g * (O[t][i] * invl) + xself[idx];
    }
  }
}

// ---------------------------------------------------------------------------
// Kernel 4: high-channel residual concat + s = out1+out2 -> sh[b][n][c] (f16)
// ---------------------------------------------------------------------------
__global__ __launch_bounds__(256) void fuse_kernel(
    const float* __restrict__ x1, const float* __restrict__ x2,
    const float* __restrict__ gammap,
    float* __restrict__ out1, float* __restrict__ out2,
    _Float16* __restrict__ sh) {
  __shared__ float ts[CIN][16];
  const int b = blockIdx.y, n0 = blockIdx.x * 16, t = threadIdx.x;
  const float g = gammap[0];
  const float* xs1 = x1 + (size_t)b * CIN * NPIX;
  const float* xs2 = x2 + (size_t)b * CIN * NPIX;
  float* o1 = out1 + (size_t)b * CIN * NPIX;
  float* o2 = out2 + (size_t)b * CIN * NPIX;
#pragma unroll 2
  for (int it = 0; it < 16; ++it) {
    int flat = t + 256 * it;
    int c = flat >> 4, nk = flat & 15;
    size_t idx = (size_t)c * NPIX + n0 + nk;
    float v1, v2;
    if (c < 128) {
      v1 = o1[idx]; v2 = o2[idx];
    } else {
      size_t lo = (size_t)(c - 128) * NPIX + n0 + nk;
      v1 = g * xs1[lo] + xs1[idx]; o1[idx] = v1;
      v2 = g * xs2[lo] + xs2[idx]; o2[idx] = v2;
    }
    ts[c][nk] = v1 + v2;
  }
  __syncthreads();
  const int nl = t >> 4, cb = t & 15;
#pragma unroll 2
  for (int it = 0; it < 16; ++it) {
    int c = cb + it * 16;
    sh[((size_t)b * NPIX + n0 + nl) * CIN + c] = (_Float16)ts[c][nl];
  }
}

// ---------------------------------------------------------------------------
// Kernel 5: 3x3 conv as implicit GEMM (9 taps x 8 K-steps of WMMA) + BN + ReLU
//  wave tile: 16 out-channels x 16 pixels (one row segment)
// ---------------------------------------------------------------------------
__global__ __launch_bounds__(256) void conv_kernel(
    const _Float16* __restrict__ sh, const _Float16* __restrict__ Wh,
    const float* __restrict__ bng, const float* __restrict__ bnb,
    float* __restrict__ feat) {
  const int w = threadIdx.x >> 5, lane = threadIdx.x & 31;
  const int r = lane & 15, hi = lane >> 4;
  const int b = blockIdx.z, ob = blockIdx.y;
  const int p0 = blockIdx.x * 16;
  const int ot = ob * 8 + w;                       // 16-channel output tile
  const int h0 = p0 >> 6, w0 = p0 & 63;
  v8f acc = zero8();
#pragma unroll 1
  for (int tap = 0; tap < 9; ++tap) {
    int dh = tap / 3 - 1, dw = tap % 3 - 1;
    int hh = h0 + dh;
    int ww = w0 + r + dw;
    bool inb = (hh >= 0) & (hh < 64) & (ww >= 0) & (ww < 64);
    size_t srow = inb ? ((size_t)b * NPIX + hh * 64 + ww) * CIN
                      : (size_t)b * NPIX * CIN;    // clamped safe address
    const _Float16* Wtap = Wh + (size_t)tap * 256 * 256;
#pragma unroll
    for (int kc = 0; kc < 8; ++kc) {               // K = 256 channels, 32/step
      AFrag a, bf;
      load_afrag(a, Wtap + (size_t)(ot * 16 + r) * 256 + kc * 32, hi);
      {
        u32x4 blo = *(const u32x4*)(sh + srow + kc * 32 + hi * 16);
        u32x4 bhi = *(const u32x4*)(sh + srow + kc * 32 + hi * 16 + 8);
#pragma unroll
        for (int j = 0; j < 4; ++j) {
          bf.u[j]     = inb ? blo[j] : 0u;
          bf.u[j + 4] = inb ? bhi[j] : 0u;
        }
      }
      acc = wmma16x16x32(a.v, bf.v, acc);
    }
  }
  const float inv = 0.9999950000374997f;           // 1/sqrt(1+1e-5)
#pragma unroll
  for (int i = 0; i < 8; ++i) {
    int o = ot * 16 + i + hi * 8;
    float y = acc[i] * (bng[o] * inv) + bnb[o];
    feat[((size_t)b * 256 + o) * NPIX + p0 + r] = fmaxf(y, 0.0f);
  }
}

// ---------------------------------------------------------------------------
// Workspace layout (in _Float16 units)
// ---------------------------------------------------------------------------
static constexpr size_t QH_OFF = 0;
static constexpr size_t KH_OFF = QH_OFF + (size_t)4 * NPIX * QKPAD;
static constexpr size_t VH_OFF = KH_OFF + (size_t)4 * NPIX * QKPAD;
static constexpr size_t SH_OFF = VH_OFF + (size_t)4 * DV * NPIX;
static constexpr size_t WH_OFF = SH_OFF + (size_t)2 * NPIX * CIN;

extern "C" void kernel_launch(void* const* d_in, const int* in_sizes, int n_in,
                              void* d_out, int out_size, void* d_ws, size_t ws_size,
                              hipStream_t stream) {
  const float* x1    = (const float*)d_in[0];
  const float* x2    = (const float*)d_in[1];
  const float* Wq    = (const float*)d_in[2];
  const float* bq    = (const float*)d_in[3];
  const float* Wk    = (const float*)d_in[4];
  const float* bk    = (const float*)d_in[5];
  const float* Wv    = (const float*)d_in[6];
  const float* bv    = (const float*)d_in[7];
  const float* gamma = (const float*)d_in[8];
  const float* Wcat  = (const float*)d_in[9];
  const float* bng   = (const float*)d_in[10];
  const float* bnb   = (const float*)d_in[11];

  float* feat = (float*)d_out;
  float* out1 = feat + (size_t)2 * CIN * NPIX;
  float* out2 = out1 + (size_t)2 * CIN * NPIX;

  _Float16* ws = (_Float16*)d_ws;
  _Float16* Qh = ws + QH_OFF;
  _Float16* Kh = ws + KH_OFF;
  _Float16* Vh = ws + VH_OFF;
  _Float16* sh = ws + SH_OFF;
  _Float16* Wh = ws + WH_OFF;

  prep_w_kernel<<<dim3((9 * 256 * 256 + 255) / 256), 256, 0, stream>>>(Wcat, Wh);
  qkv_kernel<<<dim3(NPIX / 32, 2, 2), 256, 0, stream>>>(x1, x2, Wq, bq, Wk, bk,
                                                        Wv, bv, Qh, Kh, Vh);
  attn_kernel<<<dim3(NPIX / 128, 2, 2), 256, 0, stream>>>(x1, x2, gamma, Qh, Kh,
                                                          Vh, out1, out2);
  fuse_kernel<<<dim3(NPIX / 16, 2), 256, 0, stream>>>(x1, x2, gamma, out1, out2, sh);
  conv_kernel<<<dim3(NPIX / 16, 2, 2), 256, 0, stream>>>(sh, Wh, bng, bnb, feat);
}
